// MultiheadAttention_37177236914784
// MI455X (gfx1250) — compile-verified, matrix operand order checked by reading, not run on hardware
//
#include <hip/hip_runtime.h>
#include <hip/hip_bf16.h>

// ---------------------------------------------------------------------------
// Problem constants (fixed by the reference)
// ---------------------------------------------------------------------------
#define DIM   512
#define NHEAD 8
#define HD    64
#define NB    16
#define NPG   512
#define NN    8192      // total nodes
#define NE    131072    // edges
#define NL    2         // GIN layers
#define DHID  1024

// GEMM epilogue flags
#define F_RELU    1
#define F_ACC     2
#define F_BIAS    4
#define F_STOREBF 8

typedef __attribute__((ext_vector_type(16))) __bf16 v16bf;
typedef __attribute__((ext_vector_type(8)))  float  v8f;

// CDNA5 async global->LDS copy (ASYNCcnt-tracked, see CDNA5 ISA)
#define ASYNC_LOAD_B128(ldsoff, gaddr)                                        \
    asm volatile("global_load_async_to_lds_b128 %0, %1, off"                  \
                 :: "v"(ldsoff), "v"(gaddr) : "memory")
#define WAIT_ASYNC0() asm volatile("s_wait_asynccnt 0x0" ::: "memory")

// Low 32 bits of a generic pointer to an LDS object == LDS byte address
// (ISA 10.2).
__device__ __forceinline__ unsigned lds_off(const void* p) {
    return (unsigned)(unsigned long long)p;
}

// ---------------------------------------------------------------------------
// helpers
// ---------------------------------------------------------------------------
__device__ __forceinline__ unsigned short f2bf(float x) {
    unsigned int u = __builtin_bit_cast(unsigned int, x);
    unsigned int r = u + 0x7FFFu + ((u >> 16) & 1u);   // round-to-nearest-even
    return (unsigned short)(r >> 16);
}

// A/B fragment for v_wmma_f32_16x16x32_bf16, contiguous-K source.
// Element pair p covers k0=(p/4)*16 + g*8 + 2*(p%4)  (ISA 7.12.2, 16-bit A 16x32).
__device__ __forceinline__ v16bf load_frag16(const unsigned short* rowp, int g) {
    union { v16bf v; unsigned int u[8]; } f;
#pragma unroll
    for (int p = 0; p < 8; ++p) {
        int k = ((p >> 2) << 4) | (g << 3) | ((p & 3) << 1);
        f.u[p] = *(const unsigned int*)(rowp + k);
    }
    return f.v;
}

// Same fragment but K runs along a strided axis (stride in elements).
__device__ __forceinline__ v16bf load_fragT(const unsigned short* colp, int stride, int g) {
    union { v16bf v; unsigned short s[16]; } f;
#pragma unroll
    for (int i = 0; i < 16; ++i) {
        int k = ((i >> 3) << 4) | (g << 3) | (i & 7);
        f.s[i] = colp[k * stride];
    }
    return f.v;
}

__device__ __forceinline__ v8f wmma_bf16(v16bf a, v16bf b, v8f c) {
    return __builtin_amdgcn_wmma_f32_16x16x32_bf16(false, a, false, b, (short)0, c,
                                                   false, false);
}

__device__ __forceinline__ v8f vzero8() {
    v8f z = {0.f, 0.f, 0.f, 0.f, 0.f, 0.f, 0.f, 0.f};
    return z;
}

// ---------------------------------------------------------------------------
// Tiled GEMM: C[M x N] = act(Ab[M x K] @ Bt[N x K]^T + bias) (+ C if F_ACC)
// Both operands pre-converted bf16; Bt is the weight pre-transposed to [N][K].
// block = 256 threads (8 waves), 128x128 tile, BK=64, double-buffered LDS
// filled by async global->LDS b128 copies; bf16 WMMA with f32 accumulation.
// Each wave owns a 32(M) x 64(N) sub-tile -> 16 WMMAs per K-stage.
// ---------------------------------------------------------------------------
#define BM 128
#define BN 128
#define BK 64
#define KP 8      // K padding (elements)

__global__ __launch_bounds__(256) void gemm_bf16_wmma(
    const unsigned short* __restrict__ Ab,   // [M][K] bf16
    const unsigned short* __restrict__ Bt,   // [N][K] bf16
    const float* __restrict__ bias,
    float* __restrict__ C,                   // [M][N] f32
    unsigned short* __restrict__ Cb,         // optional [M][N] bf16
    int M, int K, int Ncols, int flags)
{
    __shared__ alignas(16) unsigned short As[2][BM][BK + KP];
    __shared__ alignas(16) unsigned short Bs[2][BN][BK + KP];

    const int tid  = threadIdx.x;
    const int lane = tid & 31;
    const int wave = tid >> 5;        // 0..7
    const int wm   = wave >> 1;       // 0..3 : rows wm*32 .. +32
    const int wn   = wave & 1;        // 0..1 : cols wn*64 .. +64
    const int g    = (lane >> 4) & 1;
    const int ln   = lane & 15;
    const int m0   = blockIdx.y * BM;
    const int n0   = blockIdx.x * BN;

    const int srow = tid >> 3;        // staging row 0..31 (+32*j)
    const int scol = (tid & 7) * 8;   // staging col (8 bf16 = 16 bytes)

    auto stage = [&](int buf, int kb) {
#pragma unroll
        for (int j = 0; j < 4; ++j) {
            int row = srow + j * 32;
            unsigned long long ga = (unsigned long long)
                (Ab + (size_t)(m0 + row) * K + kb * BK + scol);
            ASYNC_LOAD_B128(lds_off(&As[buf][row][scol]), ga);
            unsigned long long gb = (unsigned long long)
                (Bt + (size_t)(n0 + row) * K + kb * BK + scol);
            ASYNC_LOAD_B128(lds_off(&Bs[buf][row][scol]), gb);
        }
    };

    v8f acc[8];
#pragma unroll
    for (int i = 0; i < 8; ++i) acc[i] = vzero8();

    const int kT = K / BK;
    stage(0, 0);
    WAIT_ASYNC0();
    __syncthreads();

    for (int kb = 0; kb < kT; ++kb) {
        const int buf = kb & 1;
        if (kb + 1 < kT) stage(buf ^ 1, kb + 1);   // overlap copy with WMMA

#pragma unroll
        for (int ks = 0; ks < 2; ++ks) {
            v16bf a0 = load_frag16(&As[buf][wm * 32 + ln][ks * 32], g);
            v16bf a1 = load_frag16(&As[buf][wm * 32 + 16 + ln][ks * 32], g);
#pragma unroll
            for (int t = 0; t < 4; ++t) {
                v16bf b = load_frag16(&Bs[buf][wn * 64 + t * 16 + ln][ks * 32], g);
                acc[t]     = wmma_bf16(a0, b, acc[t]);
                acc[4 + t] = wmma_bf16(a1, b, acc[4 + t]);
            }
        }
        WAIT_ASYNC0();        // our async writes into buf^1 have landed
        __syncthreads();      // everyone done reading buf
    }

    // epilogue
#pragma unroll
    for (int mi = 0; mi < 2; ++mi) {
#pragma unroll
        for (int t = 0; t < 4; ++t) {
            int n = n0 + wn * 64 + t * 16 + ln;
            float bv = (flags & F_BIAS) ? bias[n] : 0.f;
            const v8f a = acc[mi * 4 + t];
#pragma unroll
            for (int r = 0; r < 8; ++r) {
                int m = m0 + wm * 32 + mi * 16 + g * 8 + r;
                float v = a[r] + bv;
                if (flags & F_RELU) v = v > 0.f ? v : 0.f;
                size_t off = (size_t)m * Ncols + n;
                if (flags & F_ACC) v += C[off];
                C[off] = v;
                if (flags & F_STOREBF) Cb[off] = f2bf(v);
            }
        }
    }
}

// ---------------------------------------------------------------------------
// elementwise f32 -> bf16 (vectorized)
// ---------------------------------------------------------------------------
__global__ void conv_bf16(const float* __restrict__ in,
                          unsigned short* __restrict__ out, int n)
{
    int i = blockIdx.x * blockDim.x + threadIdx.x;
    if (i * 4 >= n) return;
    float4 v = ((const float4*)in)[i];
    unsigned short o[4] = { f2bf(v.x), f2bf(v.y), f2bf(v.z), f2bf(v.w) };
    *(uint2*)(out + (size_t)i * 4) = *(const uint2*)o;
}

// ---------------------------------------------------------------------------
// weight transpose + convert: Wt[n*K + k] = bf16(W[k*N + n])
// ---------------------------------------------------------------------------
__global__ void transpose_bf16(const float* __restrict__ W,
                               unsigned short* __restrict__ Wt, int K, int Ncols)
{
    int i = blockIdx.x * blockDim.x + threadIdx.x;
    if (i >= K * Ncols) return;
    int n = i / K, k = i - n * K;
    Wt[i] = f2bf(W[(size_t)k * Ncols + n]);
}

// ---------------------------------------------------------------------------
// m = h  (vector copy, float4)
// ---------------------------------------------------------------------------
__global__ void copy_f4(const float4* __restrict__ src, float4* __restrict__ dst, int n4) {
    int i = blockIdx.x * blockDim.x + threadIdx.x;
    if (i < n4) dst[i] = src[i];
}

// ---------------------------------------------------------------------------
// m[dst[e]] += h[src[e]]   (4 floats per thread, f32 global atomics)
// ---------------------------------------------------------------------------
__global__ void edge_scatter(const long long* __restrict__ src,
                             const long long* __restrict__ dst,
                             const float* __restrict__ h, float* __restrict__ m)
{
    int t = blockIdx.x * blockDim.x + threadIdx.x;
    if (t >= NE * (DIM / 4)) return;
    int e = t >> 7;
    int c = (t & 127) << 2;
    int s = (int)src[e];
    int d = (int)dst[e];
    float4 val = *(const float4*)(h + (size_t)s * DIM + c);
    float* dp = m + (size_t)d * DIM + c;
    atomicAdd(dp + 0, val.x);
    atomicAdd(dp + 1, val.y);
    atomicAdd(dp + 2, val.z);
    atomicAdd(dp + 3, val.w);
}

// ---------------------------------------------------------------------------
// batch-norm stats over rows (one block per feature column)
// ---------------------------------------------------------------------------
__global__ __launch_bounds__(256) void bn_stats(const float* __restrict__ m,
                                                float* __restrict__ mu,
                                                float* __restrict__ var)
{
    __shared__ float ss[256], sq[256];
    int c = blockIdx.x, tid = threadIdx.x;
    float s = 0.f, q = 0.f;
    for (int r = tid; r < NN; r += 256) {
        float v = m[(size_t)r * DIM + c];
        s += v; q += v * v;
    }
    ss[tid] = s; sq[tid] = q;
    __syncthreads();
    for (int st = 128; st > 0; st >>= 1) {
        if (tid < st) { ss[tid] += ss[tid + st]; sq[tid] += sq[tid + st]; }
        __syncthreads();
    }
    if (tid == 0) {
        float mean = ss[0] * (1.f / NN);
        mu[c]  = mean;
        var[c] = sq[0] * (1.f / NN) - mean * mean;
    }
}

// normalize in place (f32), write relu(f32) and normalized bf16 copies
__global__ void bn_apply(float* __restrict__ m, float* __restrict__ hout,
                         unsigned short* __restrict__ mb,
                         const float* __restrict__ mu, const float* __restrict__ var,
                         const float* __restrict__ gamma, const float* __restrict__ beta)
{
    int i = blockIdx.x * blockDim.x + threadIdx.x;
    if (i >= NN * DIM) return;
    int c = i & (DIM - 1);
    float inv = rsqrtf(var[c] + 1e-5f);
    float v = (m[i] - mu[c]) * inv * gamma[c] + beta[c];
    m[i] = v;
    mb[i] = f2bf(v);
    hout[i] = v > 0.f ? v : 0.f;
}

// ---------------------------------------------------------------------------
// RoPE on q,k + bf16 head-major pack of q,k,v:  out[(b*H+h)*NPG + p][64]
// ---------------------------------------------------------------------------
__global__ void rope_pack(const float* __restrict__ q, const float* __restrict__ k,
                          const float* __restrict__ v,
                          unsigned short* __restrict__ qh,
                          unsigned short* __restrict__ kh,
                          unsigned short* __restrict__ vh)
{
    int idx = blockIdx.x * blockDim.x + threadIdx.x;
    if (idx >= NN * NHEAD) return;
    int node = idx / NHEAD, h = idx % NHEAD;
    int b = node / NPG, p = node % NPG;
    int bh = b * NHEAD + h;
    const float* qr = q + (size_t)node * DIM + h * HD;
    const float* kr = k + (size_t)node * DIM + h * HD;
    const float* vr = v + (size_t)node * DIM + h * HD;
    unsigned short* qo = qh + ((size_t)bh * NPG + p) * HD;
    unsigned short* ko = kh + ((size_t)bh * NPG + p) * HD;
    unsigned short* vo = vh + ((size_t)bh * NPG + p) * HD;
    const float ln1e4_over32 = 9.210340371976184f / 32.f;
#pragma unroll 4
    for (int j = 0; j < HD / 2; ++j) {
        float invf = __expf(-(float)j * ln1e4_over32);  // 10000^(-2j/64)
        float fr = (float)p * invf;
        float sn, cs;
        __sincosf(fr, &sn, &cs);
        float q1 = qr[j], q2 = qr[j + 32];
        qo[j]      = f2bf(q1 * cs - q2 * sn);
        qo[j + 32] = f2bf(q2 * cs + q1 * sn);
        float k1 = kr[j], k2 = kr[j + 32];
        ko[j]      = f2bf(k1 * cs - k2 * sn);
        ko[j + 32] = f2bf(k2 * cs + k1 * sn);
        vo[j]      = f2bf(vr[j]);
        vo[j + 32] = f2bf(vr[j + 32]);
    }
}

// ---------------------------------------------------------------------------
// Attention: block = (bh, 128-query tile); 8 waves x 16 queries each.
// Two-pass softmax, WMMA for QK^T and P.V, P re-fragmented through LDS,
// K/V chunks staged with async global->LDS copies. Output bf16 node-major.
// ---------------------------------------------------------------------------
__global__ __launch_bounds__(256) void attn_wmma(
    const unsigned short* __restrict__ qh,
    const unsigned short* __restrict__ kh,
    const unsigned short* __restrict__ vh,
    unsigned short* __restrict__ attb)     // [NN][DIM] bf16
{
    __shared__ alignas(16) unsigned short Ks[32][HD + 8];
    __shared__ alignas(16) unsigned short Vs[32][HD + 8];
    __shared__ alignas(16) unsigned short Ps[8][16][40];

    const int bh   = blockIdx.x;
    const int qb   = blockIdx.y;
    const int tid  = threadIdx.x;
    const int lane = tid & 31;
    const int w    = tid >> 5;
    const int g    = (lane >> 4) & 1;
    const int ln   = lane & 15;
    const int qrow0 = qb * 128 + w * 16;
    const size_t base = (size_t)bh * NPG * HD;

    const int key  = tid >> 3;
    const int dim0 = (tid & 7) * 8;

    v16bf qa0, qa1;
    {
        const unsigned short* qp = qh + base + (size_t)(qrow0 + ln) * HD;
        qa0 = load_frag16(qp, g);
        qa1 = load_frag16(qp + 32, g);
    }

    float rmax[8], rsum[8];
#pragma unroll
    for (int r = 0; r < 8; ++r) { rmax[r] = -1e30f; rsum[r] = 0.f; }

    const float scale = 0.125f;           // 1/sqrt(64)

    // ---- pass 1: row max + sum(exp) ----
    for (int kc = 0; kc < NPG / 32; ++kc) {
        __syncthreads();
        ASYNC_LOAD_B128(lds_off(&Ks[key][dim0]),
            (unsigned long long)(kh + base + (size_t)(kc * 32 + key) * HD + dim0));
        WAIT_ASYNC0();
        __syncthreads();
#pragma unroll
        for (int s = 0; s < 2; ++s) {
            const unsigned short* kp = &Ks[s * 16 + ln][0];
            v8f sa = vzero8();
            sa = wmma_bf16(qa0, load_frag16(kp, g), sa);
            sa = wmma_bf16(qa1, load_frag16(kp + 32, g), sa);
#pragma unroll
            for (int r = 0; r < 8; ++r) {
                float sv = sa[r] * scale;
                float cm = sv;
                for (int off = 8; off >= 1; off >>= 1)
                    cm = fmaxf(cm, __shfl_xor(cm, off, 32));
                float nm = fmaxf(rmax[r], cm);
                float ex = __expf(sv - nm);
                for (int off = 8; off >= 1; off >>= 1)
                    ex += __shfl_xor(ex, off, 32);
                rsum[r] = rsum[r] * __expf(rmax[r] - nm) + ex;
                rmax[r] = nm;
            }
        }
    }

    // ---- pass 2: O = sum exp(S - rmax) * V ----
    v8f oacc[4];
#pragma unroll
    for (int t = 0; t < 4; ++t) oacc[t] = vzero8();

    for (int kc = 0; kc < NPG / 32; ++kc) {
        __syncthreads();
        ASYNC_LOAD_B128(lds_off(&Ks[key][dim0]),
            (unsigned long long)(kh + base + (size_t)(kc * 32 + key) * HD + dim0));
        ASYNC_LOAD_B128(lds_off(&Vs[key][dim0]),
            (unsigned long long)(vh + base + (size_t)(kc * 32 + key) * HD + dim0));
        WAIT_ASYNC0();
        __syncthreads();
#pragma unroll
        for (int s = 0; s < 2; ++s) {
            const unsigned short* kp = &Ks[s * 16 + ln][0];
            v8f sa = vzero8();
            sa = wmma_bf16(qa0, load_frag16(kp, g), sa);
            sa = wmma_bf16(qa1, load_frag16(kp + 32, g), sa);
#pragma unroll
            for (int r = 0; r < 8; ++r) {
                float p = __expf(sa[r] * scale - rmax[r]);
                Ps[w][g * 8 + r][s * 16 + ln] = f2bf(p);   // wave-private tile
            }
        }
        // wave-private LDS RAW: DS ops are in-order within a wave
        v16bf pa = load_frag16(&Ps[w][ln][0], g);
#pragma unroll
        for (int t = 0; t < 4; ++t) {
            v16bf vb = load_fragT(&Vs[0][t * 16 + ln], HD + 8, g);
            oacc[t] = wmma_bf16(pa, vb, oacc[t]);
        }
    }

    const int b = bh / NHEAD, h = bh % NHEAD;
#pragma unroll
    for (int t = 0; t < 4; ++t) {
#pragma unroll
        for (int r = 0; r < 8; ++r) {
            int qrow = qrow0 + g * 8 + r;
            int node = b * NPG + qrow;
            attb[(size_t)node * DIM + h * HD + t * 16 + ln] =
                f2bf(oacc[t][r] / rsum[r]);
        }
    }
}

// ---------------------------------------------------------------------------
// launcher
// ---------------------------------------------------------------------------
extern "C" void kernel_launch(void* const* d_in, const int* in_sizes, int n_in,
                              void* d_out, int out_size, void* d_ws, size_t ws_size,
                              hipStream_t stream)
{
    (void)in_sizes; (void)n_in; (void)out_size; (void)ws_size;

    const float*     x   = (const float*)d_in[0];
    const long long* ei  = (const long long*)d_in[1];
    const float* wq  = (const float*)d_in[3];
    const float* bq  = (const float*)d_in[4];
    const float* wk  = (const float*)d_in[5];
    const float* bk  = (const float*)d_in[6];
    const float* wv  = (const float*)d_in[7];
    const float* bv  = (const float*)d_in[8];
    const float* wqs = (const float*)d_in[9];
    const float* wks = (const float*)d_in[10];
    const float* wvs = (const float*)d_in[11];
    const float* gw1 = (const float*)d_in[12];
    const float* gb1 = (const float*)d_in[13];
    const float* gw2 = (const float*)d_in[14];
    const float* gb2 = (const float*)d_in[15];
    const float* bng = (const float*)d_in[16];
    const float* bnb = (const float*)d_in[17];
    const float* spw = (const float*)d_in[18];
    const float* wo  = (const float*)d_in[19];
    const float* bo  = (const float*)d_in[20];

    const size_t ND = (size_t)NN * DIM;
    const size_t NDH = (size_t)NN * DHID;

    float* ws    = (float*)d_ws;
    float* q     = ws;
    float* k     = q + ND;
    float* v     = k + ND;
    float* strct = v + ND;
    float* m     = strct + ND;
    float* h     = m + ND;
    float* hid   = h + ND;                       // f32 scratch, NDH floats
    float* mu    = hid + NDH;
    float* var   = mu + DIM;

    // bf16 region
    unsigned short* bfb    = (unsigned short*)(var + DIM);
    unsigned short* xb     = bfb;                // ND
    unsigned short* mb     = xb + ND;            // ND (also attention output)
    unsigned short* hidb   = mb + ND;            // NDH
    unsigned short* strctb = hidb + NDH;         // ND
    unsigned short* wT     = strctb + ND;        // transposed weights

    size_t wo_ = 0;
    auto walloc = [&](size_t sz) { unsigned short* p = wT + wo_; wo_ += sz; return p; };
    unsigned short* wqT  = walloc((size_t)DIM * DIM);
    unsigned short* wkT  = walloc((size_t)DIM * DIM);
    unsigned short* wvT  = walloc((size_t)DIM * DIM);
    unsigned short* wqsT = walloc((size_t)DIM * DIM);
    unsigned short* wksT = walloc((size_t)DIM * DIM);
    unsigned short* wvsT = walloc((size_t)DIM * DIM);
    unsigned short* woT  = walloc((size_t)DIM * DIM);
    unsigned short* gw1T[NL], *gw2T[NL], *spwT[NL];
    for (int i = 0; i < NL; ++i) {
        gw1T[i] = walloc((size_t)DIM * DHID);
        gw2T[i] = walloc((size_t)DHID * DIM);
        spwT[i] = walloc((size_t)DIM * DIM);
    }

    // head-major bf16 q/k/v overlay the f32 hid region (3*ND*2B <= NDH*4B)
    unsigned short* qhb = (unsigned short*)hid;
    unsigned short* khb = qhb + ND;
    unsigned short* vhb = khb + ND;

    dim3 blk(256);
    auto gemm = [&](const unsigned short* A, const unsigned short* Bt,
                    const float* bias, float* C, unsigned short* Cb,
                    int M, int K, int Ncols, int flags) {
        dim3 grid(Ncols / BN, M / BM);
        gemm_bf16_wmma<<<grid, blk, 0, stream>>>(A, Bt, bias, C, Cb, M, K, Ncols, flags);
    };
    auto tr = [&](const float* W, unsigned short* Wt, int K, int Ncols) {
        transpose_bf16<<<(K * Ncols + 255) / 256, blk, 0, stream>>>(W, Wt, K, Ncols);
    };

    // --- prep: convert x, transpose+convert all weights to bf16 ---
    conv_bf16<<<(unsigned)(ND / 4 / 256), blk, 0, stream>>>(x, xb, (int)ND);
    tr(wq, wqT, DIM, DIM);   tr(wk, wkT, DIM, DIM);   tr(wv, wvT, DIM, DIM);
    tr(wqs, wqsT, DIM, DIM); tr(wks, wksT, DIM, DIM); tr(wvs, wvsT, DIM, DIM);
    tr(wo, woT, DIM, DIM);
    for (int i = 0; i < NL; ++i) {
        tr(gw1 + (size_t)i * DIM * DHID, gw1T[i], DIM, DHID);
        tr(gw2 + (size_t)i * DHID * DIM, gw2T[i], DHID, DIM);
        tr(spw + (size_t)i * DIM * DIM,  spwT[i], DIM, DIM);
    }

    // --- base projections ---
    gemm(xb, wqT, bq, q, nullptr, NN, DIM, DIM, F_BIAS);
    gemm(xb, wkT, bk, k, nullptr, NN, DIM, DIM, F_BIAS);
    gemm(xb, wvT, bv, v, nullptr, NN, DIM, DIM, F_BIAS);

    // --- StructureExtractor: L GIN layers ---
    const float* hin = x;
    for (int i = 0; i < NL; ++i) {
        copy_f4<<<(unsigned)(ND / 4 / 256), blk, 0, stream>>>(
            (const float4*)hin, (float4*)m, (int)(ND / 4));
        edge_scatter<<<NE * (DIM / 4) / 256, blk, 0, stream>>>(ei, ei + NE, hin, m);
        conv_bf16<<<(unsigned)(ND / 4 / 256), blk, 0, stream>>>(m, mb, (int)ND);
        gemm(mb, gw1T[i], gb1 + i * DHID, hid, hidb,
             NN, DIM, DHID, F_BIAS | F_RELU | F_STOREBF);
        gemm(hidb, gw2T[i], gb2 + i * DIM, m, nullptr, NN, DHID, DIM, F_BIAS);
        bn_stats<<<DIM, blk, 0, stream>>>(m, mu, var);
        bn_apply<<<(unsigned)(ND / 256), blk, 0, stream>>>(
            m, h, mb, mu, var, bng + i * DIM, bnb + i * DIM);
        gemm(mb, spwT[i], nullptr, strct, (i ? strctb : nullptr),
             NN, DIM, DIM, i ? (F_ACC | F_STOREBF) : 0);
        hin = h;
    }

    // --- q/k/v += struct @ w{q,k,v}s ---
    gemm(strctb, wqsT, nullptr, q, nullptr, NN, DIM, DIM, F_ACC);
    gemm(strctb, wksT, nullptr, k, nullptr, NN, DIM, DIM, F_ACC);
    gemm(strctb, wvsT, nullptr, v, nullptr, NN, DIM, DIM, F_ACC);

    // --- RoPE + head-major pack (overwrites hid region; GIN is done) ---
    rope_pack<<<(NN * NHEAD) / 256, blk, 0, stream>>>(q, k, v, qhb, khb, vhb);

    // --- attention -> bf16 node-major (into mb) ---
    attn_wmma<<<dim3(NB * NHEAD, NPG / 128), blk, 0, stream>>>(qhb, khb, vhb, mb);

    // --- output projection ---
    gemm(mb, woT, bo, (float*)d_out, nullptr, NN, DIM, DIM, F_BIAS);
}
